// LigerMultiTokenAttention_15264313770553
// MI455X (gfx1250) — compile-verified, matrix operand order checked by reading, not run on hardware
//
#include <hip/hip_runtime.h>
#include <math.h>

typedef __attribute__((ext_vector_type(16))) _Float16 v16h;
typedef __attribute__((ext_vector_type(8)))  _Float16 v8h;
typedef __attribute__((ext_vector_type(8)))  float    v8f;

#define BB    4
#define CC    16
#define LL    1024
#define PPAD  1028      // L + 2*PAD
#define NTAP  25
#define NPAIR 13        // ceil(25/2) tap pairs -> K=32 per WMMA

// ---------------------------------------------------------------------------
// Kernel 1: zero the padded NHWC f16 probs buffer (borders + masked region).
// ---------------------------------------------------------------------------
__global__ void mta_fill_zero(uint4* __restrict__ p, size_t n4) {
    size_t i      = (size_t)blockIdx.x * blockDim.x + threadIdx.x;
    size_t stride = (size_t)gridDim.x * blockDim.x;
    uint4 z = make_uint4(0u, 0u, 0u, 0u);
    for (; i < n4; i += stride) p[i] = z;
}

// ---------------------------------------------------------------------------
// Kernel 2: causal softmax over key dim. Block = (b,y), wave w = channel w.
// Row values live in 32 VGPRs (single global read); wave32 shuffle reduce;
// transpose via LDS; one coalesced 32KB NHWC f16 burst out.
// pp layout: pp[b][yy][xx][ci], yy=y+2, xx=x+2 (2-wide zero halo).
// ---------------------------------------------------------------------------
__global__ __launch_bounds__(512)
void mta_softmax(const float* __restrict__ scores, _Float16* __restrict__ pp) {
    const int blk  = blockIdx.x;          // b*L + y
    const int b    = blk >> 10;
    const int y    = blk & (LL - 1);
    const int ci   = threadIdx.x >> 5;    // wave id = channel
    const int lane = threadIdx.x & 31;

    const float* row = scores + ((((size_t)(b * CC + ci)) * LL + y) * LL);

    float v[32];
    float m = -INFINITY;
#pragma unroll
    for (int i = 0; i < 32; ++i) {
        const int x = lane + i * 32;
        const float s = row[x];
        v[i] = s;
        if (x <= y) m = fmaxf(m, s);
    }
#pragma unroll
    for (int off = 16; off >= 1; off >>= 1)
        m = fmaxf(m, __shfl_xor(m, off, 32));

    float sum = 0.0f;
#pragma unroll
    for (int i = 0; i < 32; ++i) {
        const int x = lane + i * 32;
        const float e = (x <= y) ? __expf(v[i] - m) : 0.0f;
        v[i] = e;
        sum += e;
    }
#pragma unroll
    for (int off = 16; off >= 1; off >>= 1)
        sum += __shfl_xor(sum, off, 32);
    const float inv = 1.0f / sum;   // at least one valid key (x = 0 <= y)

    __shared__ __align__(16) _Float16 tile[LL * CC];   // 32 KB, [x][ci]
#pragma unroll
    for (int i = 0; i < 32; ++i) {
        const int x = lane + i * 32;
        tile[x * CC + ci] = (_Float16)(v[i] * inv);
    }
    __syncthreads();

    // contiguous NHWC destination: pp[b][y+2][2..1025][0..15]
    _Float16* dst = pp + ((((size_t)b * PPAD + (y + 2)) * PPAD) + 2) * CC;
    const uint4* s4 = (const uint4*)tile;
    uint4*       d4 = (uint4*)dst;
#pragma unroll
    for (int i = 0; i < 4; ++i) {
        const int idx = (int)threadIdx.x + i * 512;    // 2048 x 16B = 32KB
        d4[idx] = s4[idx];
    }
}

// ---------------------------------------------------------------------------
// Kernel 3: 5x5 conv as implicit GEMM on v_wmma_f32_16x16x32_f16.
//   N = 16 out-channels, M = 16 x-positions, K = 16 ci * 2 taps.
// One wave per (b,y) strip; walks only causal x-tiles; B fragments preloaded
// to registers (per ISA A/B 16-bit layouts, lane (n,hi): elem e -> tap
// t0+(e>=8), ci = 8*hi + (e&7) for BOTH A and B). C/D f32 layout: lane holds
// column co=n, rows x0+8*hi+r -> two contiguous float4 stores per lane.
// ---------------------------------------------------------------------------
__global__ __launch_bounds__(32)
void mta_conv_wmma(const float* __restrict__ w, const float* __restrict__ bias,
                   const _Float16* __restrict__ pp, float* __restrict__ out) {
    const int blk  = blockIdx.x;          // b*L + y
    const int b    = blk >> 10;
    const int y    = blk & (LL - 1);
    const int lane = (int)threadIdx.x & 31;
    const int n    = lane & 15;           // co (B/D column) == x-offset m (A row)
    const int hi   = lane >> 4;
    const int cib  = hi * 8;              // this lane's ci sub-range base

    // ---- preload 13 B fragments (weights, f32 -> f16), register-resident ----
    v16h bw[NPAIR];
#pragma unroll
    for (int p = 0; p < NPAIR; ++p) {
        const int t0 = 2 * p;
#pragma unroll
        for (int e = 0; e < 16; ++e) {
            const int t  = t0 + (e >= 8 ? 1 : 0);
            const int c  = cib + (e & 7);
            const float wv = (t < NTAP) ? w[(n * CC + c) * NTAP + t] : 0.0f;
            bw[p][e] = (_Float16)wv;
        }
    }
    const float bias_n = bias[n];

    const size_t rowbase = (((size_t)b * PPAD + y) * PPAD) * CC;  // padded (y,0)
    const int xtmax = y >> 4;             // last causal x-tile

    for (int xt = 0; xt <= xtmax; ++xt) {
        const int x0 = xt * 16;

        if (xt < xtmax)  // pull next tile's probs toward the caches
            __builtin_prefetch(pp + rowbase + (size_t)(x0 + 16 + n) * CC, 0, 3);

        v8f c;
#pragma unroll
        for (int r = 0; r < 8; ++r) c[r] = bias_n;

#pragma unroll
        for (int p = 0; p < NPAIR; ++p) {
            const int t0 = 2 * p, t1 = 2 * p + 1;
            const int dy0 = t0 / 5, dx0 = t0 % 5;
            const int dy1 = t1 / 5, dx1 = t1 % 5;

            const v8h a0 = *(const v8h*)(pp + rowbase +
                                ((size_t)dy0 * PPAD + x0 + n + dx0) * CC + cib);
            v8h a1;
            if (t1 < NTAP) {
                a1 = *(const v8h*)(pp + rowbase +
                                ((size_t)dy1 * PPAD + x0 + n + dx1) * CC + cib);
            } else {
#pragma unroll
                for (int e = 0; e < 8; ++e) a1[e] = (_Float16)0.0f;
            }
            union { v16h v; struct { v8h lo; v8h hi; } s; } ua;
            ua.s.lo = a0; ua.s.hi = a1;

            c = __builtin_amdgcn_wmma_f32_16x16x32_f16(
                    false, ua.v, false, bw[p], (short)0, c, false, false);
        }

        // ---- masked store: lane writes co=n, x = x0 + 8*hi + r ----
        float vals[8];
#pragma unroll
        for (int r = 0; r < 8; ++r) {
            const int x = x0 + hi * 8 + r;
            vals[r] = (x <= y) ? c[r] : 0.0f;
        }
        float* o = out + ((((size_t)b * CC + n) * LL) + y) * LL + x0 + hi * 8;
        *(float4*)(o)     = make_float4(vals[0], vals[1], vals[2], vals[3]);
        *(float4*)(o + 4) = make_float4(vals[4], vals[5], vals[6], vals[7]);
    }

    // ---- zero the strictly non-causal tail (single write, no extra pass) ----
    const int xstart = (xtmax + 1) * 16;
    for (int co = 0; co < CC; ++co) {
        float* o = out + ((((size_t)b * CC + co) * LL) + y) * LL;
        for (int x = xstart + lane; x < LL; x += 32) o[x] = 0.0f;
    }
}

// ---------------------------------------------------------------------------
extern "C" void kernel_launch(void* const* d_in, const int* in_sizes, int n_in,
                              void* d_out, int out_size, void* d_ws, size_t ws_size,
                              hipStream_t stream) {
    const float* scores = (const float*)d_in[0];   // [4,16,1024,1024] f32
    const float* weight = (const float*)d_in[1];   // [16,16,5,5] f32
    const float* bias   = (const float*)d_in[2];   // [16] f32
    float*       out    = (float*)d_out;           // [4,16,1024,1024] f32
    _Float16*    pp     = (_Float16*)d_ws;         // padded NHWC f16 probs

    const size_t ppBytes = (size_t)BB * PPAD * PPAD * CC * sizeof(_Float16);
    const size_t n4 = ppBytes / sizeof(uint4);

    mta_fill_zero<<<4096, 256, 0, stream>>>((uint4*)d_ws, n4);
    mta_softmax  <<<BB * LL, 512, 0, stream>>>(scores, pp);
    mta_conv_wmma<<<BB * LL, 32, 0, stream>>>(weight, bias, pp, out);
}
